// RPN_38001870635249
// MI455X (gfx1250) — matrix operand order repaired and checked by reading, not compiled, not run on hardware
//
#include <hip/hip_runtime.h>
#include <hip/hip_bf16.h>
#include <stdint.h>

typedef __attribute__((ext_vector_type(16))) __bf16 v16bf;
typedef __attribute__((ext_vector_type(8)))  __bf16 v8bf;
typedef __attribute__((ext_vector_type(8)))  float  v8f;

#define CIN   512
#define CO    512
#define HW    4096          // 64*64
#define KTOT  4608          // 9*512, k = (kh*3+kw)*512 + c
#define BATCH 8
#define NPROP 36864         // 9*64*64
#define NSORT 65536
#define PRE_TOPN  3000
#define POST_TOPN 300
#define NMS_TH 0.7f

__constant__ float ANCH[9][4] = {
  {-83.f,-39.f,100.f,56.f},  {-175.f,-87.f,192.f,104.f}, {-359.f,-183.f,376.f,200.f},
  {-55.f,-55.f,72.f,72.f},   {-119.f,-119.f,136.f,136.f},{-247.f,-247.f,264.f,264.f},
  {-35.f,-79.f,52.f,96.f},   {-79.f,-167.f,96.f,184.f},  {-167.f,-343.f,184.f,360.f}};

__device__ __forceinline__ unsigned pack_bf16(float lo, float hi) {
  unsigned short a = __builtin_bit_cast(unsigned short, (__bf16)lo);
  unsigned short b = __builtin_bit_cast(unsigned short, (__bf16)hi);
  return ((unsigned)b << 16) | (unsigned)a;
}

// ---------------------------------------------------------------------------
// 3x3 SAME conv as implicit GEMM, K reordered as ((kh,kw), c) so each 32-wide
// K-chunk has a uniform (scalar) filter tap. Per batch: M=4096, N=512, K=4608.
// Block: 256 threads (8 waves), tile 64(M) x 128(N), K-step 32, bf16 WMMA.
// A tile = one image row h, all 64 w, 32 channels -> coalesced loads.
// B stored transposed in LDS so both fragment types load as contiguous b128.
// ---------------------------------------------------------------------------
__global__ __launch_bounds__(256)
void conv3x3_wmma(const float* __restrict__ x, const float* __restrict__ Wc,
                  const float* __restrict__ bc, float* __restrict__ conv)
{
  __shared__ __bf16 As[64][40];    // [row=w][kk]  row stride 80B (16B aligned)
  __shared__ __bf16 Bt[128][40];   // [n][kk]      row stride 80B (16B aligned)

  const int nt  = CO / 128;                   // 4 N-tiles
  const int mt  = HW / 64;                    // 64 M-tiles (one per image row)
  const int b   = blockIdx.x / (mt * nt);
  const int rem = blockIdx.x % (mt * nt);
  const int m0  = (rem / nt) * 64;
  const int n0  = (rem % nt) * 128;
  const int h   = m0 >> 6;                    // fixed image row for this block

  const int t     = threadIdx.x;
  const int lane  = t & 31;
  const int wv    = t >> 5;
  const int waveM = wv >> 2;                  // 0..1
  const int waveN = wv & 3;                   // 0..3
  const bool hi   = lane >= 16;
  const int  ln   = lane & 15;
  const int  h8   = hi ? 1 : 0;               // A-frag v8bf chunk select
  const int  k8   = hi ? 2 : 0;               // unused; B-frag chunks also use h8

  v8f acc[2][2];
  #pragma unroll
  for (int i = 0; i < 2; i++)
    #pragma unroll
    for (int j = 0; j < 2; j++)
      #pragma unroll
      for (int e = 0; e < 8; e++) acc[i][j][e] = 0.f;

  const float* xb = x + (size_t)b * CIN * HW;
  const int row = t & 63;                     // A staging: row = w
  const int kpA = t >> 6;                     // A staging: base k-pair

  for (int k0 = 0; k0 < KTOT; k0 += 32) {
    // scalar (uniform) per step: filter tap + channel base
    const int r9 = k0 >> 9;                   // kh*3+kw, 0..8
    const int c0 = k0 & 511;
    const int dh = r9 / 3 - 1;
    const int dw = r9 - (r9 / 3) * 3 - 1;
    const int hh = h + dh;
    const bool hok = (unsigned)hh < 64u;
    const int hhc = min(max(hh, 0), 63);

    __syncthreads();
    // ---- A tile: 64 w x 32 c, pair-packed bf16, coalesced b32 loads ----
    {
      const int ww  = row + dw;
      const bool ok = hok && ((unsigned)ww < 64u);
      const int wwc = min(max(ww, 0), 63);
      const size_t base = ((size_t)(hhc << 6)) + wwc;
      #pragma unroll
      for (int i = 0; i < 4; i++) {
        int kp = kpA + i * 4;                 // k-pair 0..15
        int c  = c0 + 2 * kp;
        float f0 = xb[((size_t)c << 12) + base];
        float f1 = xb[((size_t)(c + 1) << 12) + base];
        f0 = ok ? f0 : 0.f;
        f1 = ok ? f1 : 0.f;
        ((unsigned*)&As[row][0])[kp] = pack_bf16(f0, f1);
      }
    }
    // ---- B tile (transposed): Bt[n][kk] = Wc[(n0+n)*4608 + c*9 + r9] ----
    #pragma unroll
    for (int i = 0; i < 8; i++) {
      int e  = i * 256 + t;
      int n  = e >> 4;
      int kp = e & 15;
      const float* g = Wc + (size_t)(n0 + n) * KTOT + (size_t)(c0 + 2 * kp) * 9 + r9;
      ((unsigned*)&Bt[n][0])[kp] = pack_bf16(g[0], g[9]);
    }
    __syncthreads();

    // ---- fragments: contiguous 16B chunks per CDNA5 16-bit layouts ----
    v16bf afrag[2], bfrag[2];
    #pragma unroll
    for (int tm = 0; tm < 2; tm++) {
      const v8bf* ar = (const v8bf*)&As[waveM * 32 + tm * 16 + ln][0];
      v8bf lo = ar[h8], hh8 = ar[2 + h8];
      afrag[tm] = __builtin_shufflevector(lo, hh8, 0,1,2,3,4,5,6,7,8,9,10,11,12,13,14,15);
    }
    #pragma unroll
    for (int tn = 0; tn < 2; tn++) {
      const v8bf* br = (const v8bf*)&Bt[waveN * 32 + tn * 16 + ln][0];
      v8bf lo = br[h8], hh8 = br[h8 + 2];
      bfrag[tn] = __builtin_shufflevector(lo, hh8, 0,1,2,3,4,5,6,7,8,9,10,11,12,13,14,15);
    }
    #pragma unroll
    for (int tm = 0; tm < 2; tm++)
      #pragma unroll
      for (int tn = 0; tn < 2; tn++)
        acc[tm][tn] = __builtin_amdgcn_wmma_f32_16x16x32_bf16(
            false, afrag[tm], false, bfrag[tn], (short)0, acc[tm][tn],
            false, false);
  }

  // ---- store C: reg=M&7, lane>=16 => M+8, lane&15 = N; bias fused ----
  #pragma unroll
  for (int tm = 0; tm < 2; tm++) {
    #pragma unroll
    for (int tn = 0; tn < 2; tn++) {
      int n = n0 + waveN * 32 + tn * 16 + ln;
      float bias = bc[n];
      #pragma unroll
      for (int e = 0; e < 8; e++) {
        int m = m0 + waveM * 32 + tm * 16 + e + (hi ? 8 : 0);
        conv[((size_t)(b * CO + n) << 12) + m] = acc[tm][tn][e] + bias;
      }
    }
  }
}

// ---------------------------------------------------------------------------
// 1x1 heads: reg (raw conv, 36 ch) + cls (ReLU(conv), 18 ch)
// ---------------------------------------------------------------------------
__global__ __launch_bounds__(256)
void heads_kernel(const float* __restrict__ conv,
                  const float* __restrict__ Wr, const float* __restrict__ br,
                  const float* __restrict__ Wcl, const float* __restrict__ bcl,
                  float* __restrict__ reg, float* __restrict__ cls)
{
  int tid = blockIdx.x * 256 + threadIdx.x;       // exactly 8*4096*54 threads
  int o   = tid % 54;
  int pos = (tid / 54) & 4095;
  int b   = tid / (54 * HW);
  bool isReg = o < 36;
  const float* wrow = isReg ? (Wr + o * CIN) : (Wcl + (o - 36) * CIN);
  const float* cb   = conv + ((size_t)(b * CIN) << 12) + pos;
  float acc = 0.f;
  for (int c = 0; c < CIN; c++) {
    float a = cb[(size_t)c << 12];
    if (!isReg) a = fmaxf(a, 0.f);
    acc = fmaf(wrow[c], a, acc);
  }
  if (isReg) reg[((size_t)(b * 36 + o) << 12) + pos]        = acc + br[o];
  else       cls[((size_t)(b * 18 + (o - 36)) << 12) + pos] = acc + bcl[o - 36];
}

// ---------------------------------------------------------------------------
// Scores (pair-softmax == sigmoid), anchor decode + clip, sort-key packing.
// score index i = a*4096 + h*64 + w ; proposal index p = (h*64+w)*9 + a
// (faithfully reproduces the reference's mismatched orderings).
// ---------------------------------------------------------------------------
__global__ void decode_kernel(const float* __restrict__ reg,
                              const float* __restrict__ cls,
                              const float* __restrict__ im_size,
                              float* __restrict__ scores,
                              float* __restrict__ props,
                              unsigned long long* __restrict__ keys)
{
  int tid = blockIdx.x * blockDim.x + threadIdx.x;  // 8*65536 threads
  int b = tid >> 16;
  int i = tid & 65535;
  if (i >= NPROP) { keys[tid] = 0ull; return; }     // sort padding
  int a  = i >> 12;
  int hw = i & 4095;
  int h  = hw >> 6, w = hw & 63;

  float c0 = cls[((size_t)(b * 18 + a) << 12) + hw];
  float c1 = cls[((size_t)(b * 18 + a + 9) << 12) + hw];
  float s  = 1.f / (1.f + expf(c1 - c0));           // softmax pair, class 0
  scores[(size_t)b * NPROP + i] = s;

  unsigned int fb = __float_as_uint(s);
  fb = (fb & 0x80000000u) ? ~fb : (fb | 0x80000000u);   // monotone f32->u32
  keys[tid] = ((unsigned long long)fb << 32) |
              (unsigned long long)(0xFFFFFFFFu - (unsigned)i);

  float dx = reg[((size_t)(b * 36 + 4 * a + 0) << 12) + hw];
  float dy = reg[((size_t)(b * 36 + 4 * a + 1) << 12) + hw];
  float dw = reg[((size_t)(b * 36 + 4 * a + 2) << 12) + hw];
  float dh = reg[((size_t)(b * 36 + 4 * a + 3) << 12) + hw];

  float aw = ANCH[a][2] - ANCH[a][0] + 1.f;
  float ah = ANCH[a][3] - ANCH[a][1] + 1.f;
  float cx = (float)(w * 16) + ANCH[a][0] + 0.5f * aw;
  float cy = (float)(h * 16) + ANCH[a][1] + 0.5f * ah;
  float pcx = dx * cx, pcy = dy * cy;
  float pw = expf(dw) * aw, ph = expf(dh) * ah;
  float xmax = im_size[b * 2 + 1] - 1.f;
  float ymax = im_size[b * 2 + 0] - 1.f;
  float x1 = fminf(fmaxf(pcx - 0.5f * pw, 0.f), xmax);
  float y1 = fminf(fmaxf(pcy - 0.5f * ph, 0.f), ymax);
  float x2 = fminf(fmaxf(pcx + 0.5f * pw, 0.f), xmax);
  float y2 = fminf(fmaxf(pcy + 0.5f * ph, 0.f), ymax);

  int p = hw * 9 + a;
  float* o = props + ((size_t)b * NPROP + p) * 4;
  o[0] = x1; o[1] = y1; o[2] = x2; o[3] = y2;
}

// ---------------------------------------------------------------------------
// Global bitonic sort, descending, 65536 keys per batch.
// ---------------------------------------------------------------------------
__global__ void bitonic_kernel(unsigned long long* __restrict__ keys, int j, int k)
{
  int tid = blockIdx.x * blockDim.x + threadIdx.x;  // 8*65536 threads
  int b = tid >> 16;
  int i = tid & 65535;
  int ixj = i ^ j;
  if (ixj > i) {
    unsigned long long* base = keys + ((size_t)b << 16);
    unsigned long long va = base[i], vb = base[ixj];
    bool down = (i & k) == 0;                       // descending overall
    bool sw = down ? (va < vb) : (va > vb);
    if (sw) { base[i] = vb; base[ixj] = va; }
  }
}

// ---------------------------------------------------------------------------
// Greedy NMS: 1 block/batch, 3000 sorted boxes in LDS, parallel suppression,
// serial compaction of first 300 kept, zero-padded.
// ---------------------------------------------------------------------------
__global__ __launch_bounds__(256)
void nms_kernel(const unsigned long long* __restrict__ keys,
                const float* __restrict__ props, float* __restrict__ out)
{
  __shared__ float bx1[PRE_TOPN], by1[PRE_TOPN], bx2[PRE_TOPN], by2[PRE_TOPN];
  __shared__ unsigned char keep[PRE_TOPN];
  int b = blockIdx.x;
  int t = threadIdx.x;
  const unsigned long long* kb = keys + ((size_t)b << 16);
  for (int i = t; i < PRE_TOPN; i += 256) {
    unsigned int payload = (unsigned int)(kb[i] & 0xFFFFFFFFull);
    unsigned int sidx = 0xFFFFFFFFu - payload;      // score-ordering index
    const float* p = props + ((size_t)b * NPROP + sidx) * 4;
    bx1[i] = p[0]; by1[i] = p[1]; bx2[i] = p[2]; by2[i] = p[3];
    keep[i] = 1;
  }
  __syncthreads();
  for (int i = 0; i < PRE_TOPN; i++) {
    if (keep[i]) {
      float xi1 = bx1[i], yi1 = by1[i], xi2 = bx2[i], yi2 = by2[i];
      float ai = (xi2 - xi1) * (yi2 - yi1);
      for (int j = i + 1 + t; j < PRE_TOPN; j += 256) {
        if (!keep[j]) continue;
        float xx1 = fmaxf(xi1, bx1[j]);
        float yy1 = fmaxf(yi1, by1[j]);
        float xx2 = fminf(xi2, bx2[j]);
        float yy2 = fminf(yi2, by2[j]);
        float iw = fmaxf(xx2 - xx1, 0.f);
        float ih = fmaxf(yy2 - yy1, 0.f);
        float inter = iw * ih;
        float aj = (bx2[j] - bx1[j]) * (by2[j] - by1[j]);
        float iou = inter / (ai + aj - inter + 1e-9f);
        if (iou > NMS_TH) keep[j] = 0;
      }
    }
    __syncthreads();
  }
  float* ob = out + (size_t)b * POST_TOPN * 4;
  for (int i = t; i < POST_TOPN * 4; i += 256) ob[i] = 0.f;
  __syncthreads();
  if (t == 0) {
    int cnt = 0;
    for (int i = 0; i < PRE_TOPN && cnt < POST_TOPN; i++) {
      if (keep[i]) {
        ob[cnt * 4 + 0] = bx1[i]; ob[cnt * 4 + 1] = by1[i];
        ob[cnt * 4 + 2] = bx2[i]; ob[cnt * 4 + 3] = by2[i];
        cnt++;
      }
    }
  }
}

extern "C" void kernel_launch(void* const* d_in, const int* in_sizes, int n_in,
                              void* d_out, int out_size, void* d_ws, size_t ws_size,
                              hipStream_t stream)
{
  const float* x   = (const float*)d_in[0];
  const float* im  = (const float*)d_in[1];
  const float* Wc  = (const float*)d_in[2];
  const float* bc  = (const float*)d_in[3];
  const float* Wr  = (const float*)d_in[4];
  const float* br  = (const float*)d_in[5];
  const float* Wcl = (const float*)d_in[6];
  const float* bcl = (const float*)d_in[7];

  float* out        = (float*)d_out;
  float* out_nms    = out;                                   // 8*300*4
  float* out_scores = out + BATCH * POST_TOPN * 4;           // 8*36864
  float* out_props  = out_scores + BATCH * NPROP;            // 8*36864*4

  char* ws = (char*)d_ws;
  float* conv = (float*)ws;                                              // 64 MB
  float* reg  = (float*)(ws + (size_t)67108864);                         // 4.5 MB
  float* cls  = (float*)(ws + (size_t)67108864 + 4718592);               // 2.25 MB
  unsigned long long* keys =
      (unsigned long long*)(ws + (size_t)67108864 + 4718592 + 2359296);  // 4 MB

  conv3x3_wmma<<<BATCH * (HW / 64) * (CO / 128), 256, 0, stream>>>(x, Wc, bc, conv);
  heads_kernel<<<(BATCH * HW * 54) / 256, 256, 0, stream>>>(conv, Wr, br, Wcl, bcl, reg, cls);
  decode_kernel<<<(BATCH * NSORT) / 256, 256, 0, stream>>>(reg, cls, im, out_scores, out_props, keys);
  for (int k = 2; k <= NSORT; k <<= 1)
    for (int j = k >> 1; j > 0; j >>= 1)
      bitonic_kernel<<<(BATCH * NSORT) / 256, 256, 0, stream>>>(keys, j, k);
  nms_kernel<<<BATCH, 256, 0, stream>>>(keys, out_props, out_nms);
}
